// GraphAttentionLayer_67534065762530
// MI455X (gfx1250) — matrix operand order repaired and checked by reading, not compiled, not run on hardware
//
#include <hip/hip_runtime.h>
#include <hip/hip_bf16.h>
#include <math.h>

#define DIM 128
#define NHEAD 4
#define CH 32
#define NEG_SLOPE 0.2f
#define LN_EPS 1e-5f

typedef __attribute__((ext_vector_type(16))) _Float16 v16h;
typedef __attribute__((ext_vector_type(8)))  float    v8f;

// ---------------------------------------------------------------------------
// K1: h = x @ W via v_wmma_f32_16x16x32_f16.
// One wave per 16x16 output tile; block of 256 (8 waves) covers one 16-row
// stripe across all 8 column tiles.
//
// W is staged to LDS as f16 in B-fragment order: for K-group g (16 K values),
// column n, the 16 halves a lane needs are contiguous and 32B-aligned:
//   Bl[(g*128 + n)*16 + e] = (f16) W[(g*16 + e)*128 + n]
// so each B fragment is a single contiguous v16h (two ds_load_b128).
// A rows are clamped (not predicated) so loads are unconditional float4s.
// ---------------------------------------------------------------------------
__global__ __launch_bounds__(256) void gat_gemm_wmma(
    const float* __restrict__ x, const float* __restrict__ W,
    float* __restrict__ h, int N)
{
    __shared__ __align__(32) _Float16 Bl[DIM * DIM];   // 32 KB of 320 KB LDS

    for (int i = threadIdx.x; i < DIM * DIM; i += 256) {
        int k = i >> 7;                 // i = k*128 + n
        int n = i & 127;
        int g = k >> 4, e = k & 15;
        Bl[(((g << 7) + n) << 4) + e] = (_Float16)W[i];
    }
    __syncthreads();

    const int wave  = threadIdx.x >> 5;
    const int lane  = threadIdx.x & 31;
    const int lhalf = lane & 15;
    const bool hi   = lane >= 16;
    const int m0    = blockIdx.x * 16;
    const int n0    = wave * 16;

    // Clamp instead of predicating: always-legal unconditional loads.
    const int arow = min(m0 + lhalf, N - 1);
    const float4* __restrict__ xrow = (const float4*)(x + (size_t)arow * DIM);

    v8f acc = {};
    #pragma unroll
    for (int kb = 0; kb < DIM; kb += 32) {
        // A fragment (contiguous k order, same as B): lane<16 holds M=lane,
        // K = {kb..kb+15}; lane>=16 K shifted +16.
        const int q0 = (kb >> 2) + (hi ? 4 : 0);   // float4 index of K=ka0
        float4 f0 = xrow[q0];
        float4 f1 = xrow[q0 + 1];
        float4 f2 = xrow[q0 + 2];
        float4 f3 = xrow[q0 + 3];
        v16h a;
        a[0]  = (_Float16)f0.x; a[1]  = (_Float16)f0.y;
        a[2]  = (_Float16)f0.z; a[3]  = (_Float16)f0.w;
        a[4]  = (_Float16)f1.x; a[5]  = (_Float16)f1.y;
        a[6]  = (_Float16)f1.z; a[7]  = (_Float16)f1.w;
        a[8]  = (_Float16)f2.x; a[9]  = (_Float16)f2.y;
        a[10] = (_Float16)f2.z; a[11] = (_Float16)f2.w;
        a[12] = (_Float16)f3.x; a[13] = (_Float16)f3.y;
        a[14] = (_Float16)f3.z; a[15] = (_Float16)f3.w;

        // B fragment: one contiguous 32B chunk per lane.
        const int g = (kb >> 4) + (hi ? 1 : 0);
        v16h b = *(const v16h*)&Bl[(((g << 7) + n0 + lhalf) << 4)];

        acc = __builtin_amdgcn_wmma_f32_16x16x32_f16(
            false, a, false, b, (short)0, acc, false, false);
    }

    // C/D layout: VGPR r -> row m0 + r (+8 for upper lane half), col n0+lhalf.
    float* __restrict__ hcol = h + (size_t)m0 * DIM + n0 + lhalf + (hi ? 8 * DIM : 0);
    if (m0 + 16 <= N) {                 // wave-uniform fast path (all blocks when N%16==0)
        #pragma unroll
        for (int r = 0; r < 8; ++r)
            hcol[(size_t)r * DIM] = acc[r];
    } else {
        #pragma unroll
        for (int r = 0; r < 8; ++r) {
            int row = m0 + r + (hi ? 8 : 0);
            if (row < N) h[(size_t)row * DIM + n0 + lhalf] = acc[r];
        }
    }
}

// ---------------------------------------------------------------------------
// K2: per-node, per-head attention logits. One wave32 == one head (C=32).
// ---------------------------------------------------------------------------
__global__ __launch_bounds__(128) void gat_attn_coeff(
    const float* __restrict__ h,
    const float* __restrict__ att_src, const float* __restrict__ att_dst,
    float* __restrict__ a_src, float* __restrict__ a_dst, int N)
{
    const int n    = blockIdx.x;
    const int head = threadIdx.x >> 5;
    const int lane = threadIdx.x & 31;
    const int c    = head * CH + lane;

    float hv = h[(size_t)n * DIM + c];
    float ps = hv * att_src[c];
    float pd = hv * att_dst[c];
    #pragma unroll
    for (int off = 16; off; off >>= 1) {
        ps += __shfl_down(ps, off, 32);
        pd += __shfl_down(pd, off, 32);
    }
    if (lane == 0) {
        a_src[(size_t)n * NHEAD + head] = ps;
        a_dst[(size_t)n * NHEAD + head] = pd;
    }
}

// Order-preserving float<->uint map so segment-max is a u32 atomicMax.
// Map sends every float above 0x00000000, so memset(0) == "-inf".
__device__ __forceinline__ unsigned f32_ord(float v) {
    unsigned u = __float_as_uint(v);
    return (u >> 31) ? ~u : (u | 0x80000000u);
}
__device__ __forceinline__ float ord_f32(unsigned u) {
    return (u & 0x80000000u) ? __uint_as_float(u ^ 0x80000000u)
                             : __uint_as_float(~u);
}
__device__ __forceinline__ void edge_nodes(int e, int E, const int* __restrict__ s_idx,
                                           const int* __restrict__ d_idx, int& s, int& d) {
    if (e < E) { s = s_idx[e]; d = d_idx[e]; }
    else       { s = d = e - E; }            // synthesized self-loop
}

// ---------------------------------------------------------------------------
// K3: segment max of leaky-relu logits (thread per edge, 4 heads each).
// ---------------------------------------------------------------------------
__global__ __launch_bounds__(256) void gat_edge_max(
    const int* __restrict__ s_idx, const int* __restrict__ d_idx, int E, int Etot,
    const float* __restrict__ a_src, const float* __restrict__ a_dst,
    unsigned* __restrict__ m_u)
{
    int e = blockIdx.x * blockDim.x + threadIdx.x;
    if (e >= Etot) return;
    int s, d; edge_nodes(e, E, s_idx, d_idx, s, d);
    const float4 as = *(const float4*)(a_src + (size_t)s * NHEAD);
    const float4 ad = *(const float4*)(a_dst + (size_t)d * NHEAD);
    float ev[NHEAD] = {as.x + ad.x, as.y + ad.y, as.z + ad.z, as.w + ad.w};
    #pragma unroll
    for (int hh = 0; hh < NHEAD; ++hh) {
        float v = ev[hh];
        v = v > 0.0f ? v : NEG_SLOPE * v;
        atomicMax(&m_u[(size_t)d * NHEAD + hh], f32_ord(v));
    }
}

// ---------------------------------------------------------------------------
// K4: softmax denominator (thread per edge).
// ---------------------------------------------------------------------------
__global__ __launch_bounds__(256) void gat_edge_denom(
    const int* __restrict__ s_idx, const int* __restrict__ d_idx, int E, int Etot,
    const float* __restrict__ a_src, const float* __restrict__ a_dst,
    const unsigned* __restrict__ m_u, float* __restrict__ denom)
{
    int e = blockIdx.x * blockDim.x + threadIdx.x;
    if (e >= Etot) return;
    int s, d; edge_nodes(e, E, s_idx, d_idx, s, d);
    const float4 as = *(const float4*)(a_src + (size_t)s * NHEAD);
    const float4 ad = *(const float4*)(a_dst + (size_t)d * NHEAD);
    float ev[NHEAD] = {as.x + ad.x, as.y + ad.y, as.z + ad.z, as.w + ad.w};
    #pragma unroll
    for (int hh = 0; hh < NHEAD; ++hh) {
        float v = ev[hh];
        v = v > 0.0f ? v : NEG_SLOPE * v;
        float m = ord_f32(m_u[(size_t)d * NHEAD + hh]);
        atomicAdd(&denom[(size_t)d * NHEAD + hh], __expf(v - m));
    }
}

// ---------------------------------------------------------------------------
// K5: scatter-aggregate agg[dst] += alpha * h[src]. One wave32 per edge;
// lane owns 4 channels (float4 load + 4 f32 atomics, resolving in L2 since
// h+agg ~= 51 MB << 192 MB L2).
// ---------------------------------------------------------------------------
__global__ __launch_bounds__(256) void gat_edge_aggregate(
    const int* __restrict__ s_idx, const int* __restrict__ d_idx, int E, int Etot,
    const float* __restrict__ a_src, const float* __restrict__ a_dst,
    const unsigned* __restrict__ m_u, const float* __restrict__ denom,
    const float* __restrict__ h, float* __restrict__ agg)
{
    int e    = (blockIdx.x * blockDim.x + threadIdx.x) >> 5;
    int lane = threadIdx.x & 31;
    if (e >= Etot) return;
    int s, d; edge_nodes(e, E, s_idx, d_idx, s, d);

    int head = lane >> 3;                      // channels [lane*4, lane*4+3] -> head
    float v = a_src[(size_t)s * NHEAD + head] + a_dst[(size_t)d * NHEAD + head];
    v = v > 0.0f ? v : NEG_SLOPE * v;
    float m  = ord_f32(m_u[(size_t)d * NHEAD + head]);
    float dn = denom[(size_t)d * NHEAD + head] + 1e-16f;
    float alpha = __expf(v - m) / dn;

    const float4 hv = *(const float4*)(h + (size_t)s * DIM + lane * 4);
    float* out = agg + (size_t)d * DIM + lane * 4;
    atomicAdd(out + 0, alpha * hv.x);
    atomicAdd(out + 1, alpha * hv.y);
    atomicAdd(out + 2, alpha * hv.z);
    atomicAdd(out + 3, alpha * hv.w);
}

// ---------------------------------------------------------------------------
// K6: residual + LayerNorm. One wave32 per node, lane owns 4 channels,
// mean/var via __shfl_xor full-wave reduction.
// ---------------------------------------------------------------------------
__global__ __launch_bounds__(256) void gat_residual_ln(
    const float* __restrict__ x, const float* __restrict__ agg,
    const float* __restrict__ bias, const float* __restrict__ gamma,
    const float* __restrict__ beta, float* __restrict__ out, int N)
{
    int node = (blockIdx.x * blockDim.x + threadIdx.x) >> 5;
    int lane = threadIdx.x & 31;
    if (node >= N) return;

    const float4 xv = *(const float4*)(x    + (size_t)node * DIM + lane * 4);
    const float4 av = *(const float4*)(agg  + (size_t)node * DIM + lane * 4);
    const float4 bv = *(const float4*)(bias + lane * 4);
    float y[4] = {xv.x + av.x + bv.x, xv.y + av.y + bv.y,
                  xv.z + av.z + bv.z, xv.w + av.w + bv.w};

    float s = y[0] + y[1] + y[2] + y[3];
    float s2 = y[0]*y[0] + y[1]*y[1] + y[2]*y[2] + y[3]*y[3];
    #pragma unroll
    for (int off = 16; off; off >>= 1) {
        s  += __shfl_xor(s,  off, 32);
        s2 += __shfl_xor(s2, off, 32);
    }
    float mean = s * (1.0f / DIM);
    float var  = s2 * (1.0f / DIM) - mean * mean;
    float inv  = rsqrtf(var + LN_EPS);

    const float4 gv = *(const float4*)(gamma + lane * 4);
    const float4 tv = *(const float4*)(beta  + lane * 4);
    float4 o;
    o.x = gv.x * (y[0] - mean) * inv + tv.x;
    o.y = gv.y * (y[1] - mean) * inv + tv.y;
    o.z = gv.z * (y[2] - mean) * inv + tv.z;
    o.w = gv.w * (y[3] - mean) * inv + tv.w;
    *(float4*)(out + (size_t)node * DIM + lane * 4) = o;
}

// ---------------------------------------------------------------------------
extern "C" void kernel_launch(void* const* d_in, const int* in_sizes, int n_in,
                              void* d_out, int out_size, void* d_ws, size_t ws_size,
                              hipStream_t stream)
{
    const float* x       = (const float*)d_in[0];
    const float* W       = (const float*)d_in[1];
    const float* att_src = (const float*)d_in[2];
    const float* att_dst = (const float*)d_in[3];
    const float* bias    = (const float*)d_in[4];
    const float* gamma   = (const float*)d_in[5];
    const float* beta    = (const float*)d_in[6];
    const int*   eidx    = (const int*)d_in[7];   // [2, E] flat, int32

    const int N    = in_sizes[0] / DIM;
    const int E    = in_sizes[7] / 2;
    const int Etot = E + N;
    const int* s_idx = eidx;
    const int* d_idx = eidx + E;

    // Workspace layout
    char* p = (char*)d_ws;
    float*    h      = (float*)p;    p += (size_t)N * DIM   * sizeof(float);
    float*    a_src  = (float*)p;    p += (size_t)N * NHEAD * sizeof(float);
    float*    a_dst  = (float*)p;    p += (size_t)N * NHEAD * sizeof(float);
    unsigned* m_u    = (unsigned*)p; p += (size_t)N * NHEAD * sizeof(unsigned);
    float*    denom  = (float*)p;    p += (size_t)N * NHEAD * sizeof(float);
    float*    agg    = (float*)p;    p += (size_t)N * DIM   * sizeof(float);

    // Zero m_u (ordered-map "-inf"), denom, agg in one capture-safe memset.
    size_t zero_bytes = ((size_t)N * NHEAD * 2 + (size_t)N * DIM) * sizeof(float);
    (void)hipMemsetAsync(m_u, 0, zero_bytes, stream);

    gat_gemm_wmma<<<(N + 15) / 16, 256, 0, stream>>>(x, W, h, N);
    gat_attn_coeff<<<N, 128, 0, stream>>>(h, att_src, att_dst, a_src, a_dst, N);
    gat_edge_max<<<(Etot + 255) / 256, 256, 0, stream>>>(
        s_idx, d_idx, E, Etot, a_src, a_dst, m_u);
    gat_edge_denom<<<(Etot + 255) / 256, 256, 0, stream>>>(
        s_idx, d_idx, E, Etot, a_src, a_dst, m_u, denom);
    gat_edge_aggregate<<<(Etot + 7) / 8, 256, 0, stream>>>(
        s_idx, d_idx, E, Etot, a_src, a_dst, m_u, denom, h, agg);
    gat_residual_ln<<<(N + 7) / 8, 256, 0, stream>>>(
        x, agg, bias, gamma, beta, (float*)d_out, N);
}
